// GNNLayer_60498909331523
// MI455X (gfx1250) — matrix operand order repaired and checked by reading, not compiled, not run on hardware
//
#include <hip/hip_runtime.h>
#include <hip/hip_bf16.h>

#define Bn 4
#define Tn 4096
#define Dn 128
#define Hn 4
#define DHn 32
#define FFn 256
#define Nn (Bn * Tn) // 16384 tokens
#define MW (Tn / 32) // 128 mask words per row

typedef __attribute__((ext_vector_type(16))) __bf16 v16bf;
typedef __attribute__((ext_vector_type(8))) float v8f;

union AF { v16bf v; unsigned u[8]; };
union BFr { v16bf v; unsigned u[8]; };

__device__ __forceinline__ v8f vzero8() {
  v8f c;
#pragma unroll
  for (int i = 0; i < 8; ++i) c[i] = 0.0f;
  return c;
}

__device__ __forceinline__ v8f wmma_bf16(v16bf a, v16bf b, v8f c) {
  // 8 args: (neg_a, A, neg_b, B, c_mod, C, reuse_a, reuse_b)
  return __builtin_amdgcn_wmma_f32_16x16x32_bf16(false, a, false, b, (short)0, c,
                                                 false, false);
}

// A-fragment (16x32 bf16) per-lane K offset for vgpr v, lane-half hi.
__device__ __forceinline__ int akoff(int v, int hi) {
  return ((v & 3) * 2) + ((v >> 2) * 16) + hi * 8;
}

__global__ void cvt_bf16(const float* __restrict__ s, __bf16* __restrict__ d, int n) {
  int i = blockIdx.x * blockDim.x + threadIdx.x;
  if (i < n) d[i] = (__bf16)s[i];
}

// -------- Pack (adj | eye) into bit mask: m32[i][w] bit j = allow(i, w*32+j) ---
__global__ void pack_mask(const unsigned char* __restrict__ adj,
                          unsigned* __restrict__ m32) {
  int idx = blockIdx.x * blockDim.x + threadIdx.x;  // one 32-bit word per thread
  int i = idx >> 7;   // row
  int w = idx & (MW - 1);
  const unsigned* p = (const unsigned*)(adj + (size_t)i * Tn + w * 32);
  unsigned bits = 0;
#pragma unroll
  for (int k = 0; k < 8; ++k) {
    unsigned wd = p[k];
#pragma unroll
    for (int byi = 0; byi < 4; ++byi)
      bits |= (((wd >> (8 * byi)) & 0xFFu) ? 1u : 0u) << (k * 4 + byi);
  }
  int diag = i - w * 32;
  if (diag >= 0 && diag < 32) bits |= (1u << diag);  // | eye
  m32[idx] = bits;
}

// ---------------- QKV projection: one wave per 16x16 tile of one of Q/K/V ------
__global__ void qkv_kernel(const __bf16* __restrict__ xb,
                           const __bf16* __restrict__ wq, const __bf16* __restrict__ wk,
                           const __bf16* __restrict__ wv,
                           const float* __restrict__ bq, const float* __restrict__ bk,
                           const float* __restrict__ bv,
                           __bf16* __restrict__ Qh, __bf16* __restrict__ Kt,
                           __bf16* __restrict__ Vh) {
  int wid = (blockIdx.x * blockDim.x + threadIdx.x) >> 5;
  int lane = threadIdx.x & 31;
  int mtile = wid / 24;
  int rem = wid % 24;
  int mat = rem >> 3;   // 0=Q 1=K 2=V
  int ntile = rem & 7;  // 8 x 16 output cols = 128
  const __bf16* W = (mat == 0) ? wq : (mat == 1) ? wk : wv;
  const float* bias = (mat == 0) ? bq : (mat == 1) ? bk : bv;
  int hi = lane >> 4;
  int arow = mtile * 16 + (lane & 15);
  v8f c = vzero8();
#pragma unroll
  for (int kc = 0; kc < 4; ++kc) {
    AF a; BFr bb;
#pragma unroll
    for (int v = 0; v < 8; ++v)
      a.u[v] = *(const unsigned*)(xb + (size_t)arow * Dn + kc * 32 + akoff(v, hi));
    for (int e = 0; e < 16; ++e)
      bb.v[e] = W[(size_t)(kc * 32 + akoff(e >> 1, hi) + (e & 1)) * Dn + ntile * 16 + (lane & 15)];
    c = wmma_bf16(a.v, bb.v, c);
  }
  int ncol = ntile * 16 + (lane & 15);
  int h = ncol >> 5, d = ncol & 31;
  float bsc = bias[ncol];
#pragma unroll
  for (int r = 0; r < 8; ++r) {
    int bt = mtile * 16 + hi * 8 + r;
    int b = bt >> 12, t = bt & (Tn - 1);
    float val = c[r] + bsc;
    if (mat == 0)
      Qh[((size_t)(b * Hn + h) * Tn + t) * DHn + d] = (__bf16)val;
    else if (mat == 1)
      Kt[((size_t)(b * Hn + h) * DHn + d) * Tn + t] = (__bf16)val;  // transposed
    else
      Vh[((size_t)(b * Hn + h) * Tn + t) * DHn + d] = (__bf16)val;
  }
}

// ---------------- Flash attention: one wave per (b, h, 16-query tile) ----------
__global__ void attn_kernel(const __bf16* __restrict__ Qh, const __bf16* __restrict__ Kt,
                            const __bf16* __restrict__ Vh,
                            const unsigned* __restrict__ msk,
                            __bf16* __restrict__ attnout) {
  __shared__ __align__(16) __bf16 plds[4][16 * 32];
  int wslot = threadIdx.x >> 5;
  int lane = threadIdx.x & 31;
  int wid = blockIdx.x * 4 + wslot;
  int qtile = wid & 255;
  int h = (wid >> 8) & 3;
  int b = wid >> 10;
  int qbase = qtile * 16;
  const __bf16* Qp = Qh + (size_t)(b * Hn + h) * Tn * DHn;
  const __bf16* Ktp = Kt + (size_t)(b * Hn + h) * DHn * Tn;
  const __bf16* Vp = Vh + (size_t)(b * Hn + h) * Tn * DHn;
  int hi = lane >> 4;
  int nn = lane & 15;

  AF qa;  // Q tile 16x32 = exactly one A-fragment, resident all loop long
  {
    int arow = qbase + (lane & 15);
#pragma unroll
    for (int v = 0; v < 8; ++v)
      qa.u[v] = *(const unsigned*)(Qp + (size_t)arow * DHn + akoff(v, hi));
  }
  BFr ones;  // B-frag of 1.0: rowsum(P) = P x ones via one WMMA
#pragma unroll
  for (int e = 0; e < 16; ++e) ones.v[e] = (__bf16)1.0f;

  v8f o0 = vzero8(), o1 = vzero8();
  float mrow[8], lrow[8];
#pragma unroll
  for (int r = 0; r < 8; ++r) { mrow[r] = -1e30f; lrow[r] = 0.0f; }
  const float scale = 0.17677669529663687f;  // 1/sqrt(32)

  for (int jt = 0; jt < Tn / 32; ++jt) {
    int j0 = jt * 32;
    BFr kb0, kb1;  // B-frag: lane = key j, 16 contraction d values in A-frag k order
    for (int e = 0; e < 16; ++e) {
      kb0.v[e] = Ktp[(size_t)(akoff(e >> 1, hi) + (e & 1)) * Tn + j0 + nn];
      kb1.v[e] = Ktp[(size_t)(akoff(e >> 1, hi) + (e & 1)) * Tn + j0 + 16 + nn];
    }
    BFr vb0, vb1;  // issued early: overlap with softmax VALU below
    for (int e = 0; e < 16; ++e) {
      vb0.v[e] = Vp[(size_t)(j0 + akoff(e >> 1, hi) + (e & 1)) * DHn + nn];
      vb1.v[e] = Vp[(size_t)(j0 + akoff(e >> 1, hi) + (e & 1)) * DHn + 16 + nn];
    }
    // prefetch next iteration's K/V lines (global_prefetch_b8)
    __builtin_prefetch((const void*)(Ktp + (size_t)lane * Tn + j0 + 32), 0, 3);
    __builtin_prefetch((const void*)(Vp + (size_t)(j0 + 32 + lane) * DHn), 0, 3);

    v8f s0 = wmma_bf16(qa.v, kb0.v, vzero8());
    v8f s1 = wmma_bf16(qa.v, kb1.v, vzero8());
    float alpha[8];
#pragma unroll
    for (int r = 0; r < 8; ++r) {
      int i = qbase + hi * 8 + r;
      unsigned mw = msk[(size_t)i * MW + jt];  // bit-packed (adj|eye) row word
      float a0 = ((mw >> nn) & 1u) ? s0[r] * scale : -1e30f;
      float a1 = ((mw >> (nn + 16)) & 1u) ? s1[r] * scale : -1e30f;
      float rm = fmaxf(a0, a1);  // row max across the 16 lanes holding this row
      rm = fmaxf(rm, __shfl_xor(rm, 1, 32));
      rm = fmaxf(rm, __shfl_xor(rm, 2, 32));
      rm = fmaxf(rm, __shfl_xor(rm, 4, 32));
      rm = fmaxf(rm, __shfl_xor(rm, 8, 32));
      float mo = mrow[r];
      float mn = fmaxf(mo, rm);
      alpha[r] = (mo == mn) ? 1.0f : __expf(mo - mn);
      float p0 = (a0 <= -1e29f) ? 0.0f : __expf(a0 - mn);
      float p1 = (a1 <= -1e29f) ? 0.0f : __expf(a1 - mn);
      mrow[r] = mn;
      o0[r] *= alpha[r];
      o1[r] *= alpha[r];
      int m = hi * 8 + r;  // C-frag -> LDS (row-major 16x32) for A-frag reload
      plds[wslot][m * 32 + nn] = (__bf16)p0;
      plds[wslot][m * 32 + 16 + nn] = (__bf16)p1;
    }
    AF pa;
#pragma unroll
    for (int v = 0; v < 8; ++v)
      pa.u[v] = *(const unsigned*)(&plds[wslot][(lane & 15) * 32 + akoff(v, hi)]);
    v8f rs = wmma_bf16(pa.v, ones.v, vzero8());  // rowsum via WMMA (no shuffles)
#pragma unroll
    for (int r = 0; r < 8; ++r) lrow[r] = lrow[r] * alpha[r] + rs[r];
    o0 = wmma_bf16(pa.v, vb0.v, o0);
    o1 = wmma_bf16(pa.v, vb1.v, o1);
  }
#pragma unroll
  for (int r = 0; r < 8; ++r) {
    float inv = (lrow[r] > 0.0f) ? 1.0f / lrow[r] : 0.0f;  // NaN-cleanup semantics
    size_t bt = (size_t)b * Tn + qbase + hi * 8 + r;
    attnout[bt * Dn + h * 32 + nn] = (__bf16)(o0[r] * inv);
    attnout[bt * Dn + h * 32 + 16 + nn] = (__bf16)(o1[r] * inv);
  }
}

// ------------- GEMM (N=128) + bias + residual + LayerNorm, fused --------------
template <int KC>
__global__ void gemm_ln_kernel(const __bf16* __restrict__ A, int lda,
                               const __bf16* __restrict__ W,
                               const float* __restrict__ bias,
                               const float* __restrict__ resid,
                               const float* __restrict__ g, const float* __restrict__ beta,
                               float* __restrict__ outf, __bf16* __restrict__ outb) {
  __shared__ float tile[16 * 128];
  int wslot = threadIdx.x >> 5;  // 8 waves: one 16-col stripe each
  int lane = threadIdx.x & 31;
  int mtile = blockIdx.x;
  int ntile = wslot;
  int hi = lane >> 4;
  int arow = mtile * 16 + (lane & 15);
  v8f c = vzero8();
#pragma unroll
  for (int kc = 0; kc < KC; ++kc) {
    AF a; BFr bb;
#pragma unroll
    for (int v = 0; v < 8; ++v)
      a.u[v] = *(const unsigned*)(A + (size_t)arow * lda + kc * 32 + akoff(v, hi));
    for (int e = 0; e < 16; ++e)
      bb.v[e] = W[(size_t)(kc * 32 + akoff(e >> 1, hi) + (e & 1)) * 128 + ntile * 16 + (lane & 15)];
    c = wmma_bf16(a.v, bb.v, c);
  }
  int ncol = ntile * 16 + (lane & 15);
  float bv = bias[ncol];
#pragma unroll
  for (int r = 0; r < 8; ++r) tile[(hi * 8 + r) * 128 + ncol] = c[r] + bv;
  __syncthreads();
  // LayerNorm: 16 threads per row, 8 cols each; shfl-xor reduce within 16-group
  int rrow = threadIdx.x >> 4, s = threadIdx.x & 15;
  size_t bt = (size_t)mtile * 16 + rrow;
  float vals[8], sum = 0.0f, sq = 0.0f;
#pragma unroll
  for (int e = 0; e < 8; ++e) {
    int cidx = s * 8 + e;
    float v = tile[rrow * 128 + cidx] + resid[bt * 128 + cidx];
    vals[e] = v;
    sum += v;
    sq += v * v;
  }
  sum += __shfl_xor(sum, 1, 32); sq += __shfl_xor(sq, 1, 32);
  sum += __shfl_xor(sum, 2, 32); sq += __shfl_xor(sq, 2, 32);
  sum += __shfl_xor(sum, 4, 32); sq += __shfl_xor(sq, 4, 32);
  sum += __shfl_xor(sum, 8, 32); sq += __shfl_xor(sq, 8, 32);
  float mean = sum * (1.0f / 128.0f);
  float var = sq * (1.0f / 128.0f) - mean * mean;
  float rstd = rsqrtf(var + 1e-5f);
#pragma unroll
  for (int e = 0; e < 8; ++e) {
    int cidx = s * 8 + e;
    float y = (vals[e] - mean) * rstd * g[cidx] + beta[cidx];
    if (outf) outf[bt * 128 + cidx] = y;
    if (outb) outb[bt * 128 + cidx] = (__bf16)y;
  }
}

// ---------------- FFN first GEMM + exact GELU ----------------------------------
__global__ void ff1_kernel(const __bf16* __restrict__ x1b, const __bf16* __restrict__ w1,
                           const float* __restrict__ b1, __bf16* __restrict__ ff) {
  int wid = (blockIdx.x * blockDim.x + threadIdx.x) >> 5;
  int lane = threadIdx.x & 31;
  int mtile = wid >> 4;
  int ntile = wid & 15;  // 16 x 16 = 256 cols
  int hi = lane >> 4;
  int arow = mtile * 16 + (lane & 15);
  v8f c = vzero8();
#pragma unroll
  for (int kc = 0; kc < 4; ++kc) {
    AF a; BFr bb;
#pragma unroll
    for (int v = 0; v < 8; ++v)
      a.u[v] = *(const unsigned*)(x1b + (size_t)arow * Dn + kc * 32 + akoff(v, hi));
    for (int e = 0; e < 16; ++e)
      bb.v[e] = w1[(size_t)(kc * 32 + akoff(e >> 1, hi) + (e & 1)) * FFn + ntile * 16 + (lane & 15)];
    c = wmma_bf16(a.v, bb.v, c);
  }
  int ncol = ntile * 16 + (lane & 15);
  float bv = b1[ncol];
#pragma unroll
  for (int r = 0; r < 8; ++r) {
    size_t bt = (size_t)mtile * 16 + hi * 8 + r;
    float v = c[r] + bv;
    float gel = 0.5f * v * (1.0f + erff(v * 0.70710678118654752f));  // exact gelu
    ff[bt * FFn + ncol] = (__bf16)gel;
  }
}

extern "C" void kernel_launch(void* const* d_in, const int* in_sizes, int n_in,
                              void* d_out, int out_size, void* d_ws, size_t ws_size,
                              hipStream_t stream) {
  const float* x = (const float*)d_in[0];
  const unsigned char* adj = (const unsigned char*)d_in[1];
  const float* Wq = (const float*)d_in[2];  const float* bq = (const float*)d_in[3];
  const float* Wk = (const float*)d_in[4];  const float* bk = (const float*)d_in[5];
  const float* Wv = (const float*)d_in[6];  const float* bv = (const float*)d_in[7];
  const float* Wo = (const float*)d_in[8];  const float* bo = (const float*)d_in[9];
  const float* ln1_g = (const float*)d_in[10]; const float* ln1_b = (const float*)d_in[11];
  const float* W1 = (const float*)d_in[12]; const float* b1 = (const float*)d_in[13];
  const float* W2 = (const float*)d_in[14]; const float* b2 = (const float*)d_in[15];
  const float* ln2_g = (const float*)d_in[16]; const float* ln2_b = (const float*)d_in[17];

  char* p = (char*)d_ws;
  auto alloc = [&](size_t bytes) {
    char* r = p;
    p += (bytes + 255) & ~(size_t)255;
    return r;
  };
  __bf16* xb  = (__bf16*)alloc((size_t)Nn * Dn * 2);
  __bf16* wqb = (__bf16*)alloc(128 * 128 * 2);
  __bf16* wkb = (__bf16*)alloc(128 * 128 * 2);
  __bf16* wvb = (__bf16*)alloc(128 * 128 * 2);
  __bf16* wob = (__bf16*)alloc(128 * 128 * 2);
  __bf16* w1b = (__bf16*)alloc(128 * 256 * 2);
  __bf16* w2b = (__bf16*)alloc(256 * 128 * 2);
  __bf16* Qh  = (__bf16*)alloc((size_t)Nn * Dn * 2);
  __bf16* Kt  = (__bf16*)alloc((size_t)Nn * Dn * 2);
  __bf16* Vh  = (__bf16*)alloc((size_t)Nn * Dn * 2);
  __bf16* att = (__bf16*)alloc((size_t)Nn * Dn * 2);
  float*  x1  = (float*) alloc((size_t)Nn * Dn * 4);
  __bf16* x1b = (__bf16*)alloc((size_t)Nn * Dn * 2);
  __bf16* ffb = (__bf16*)alloc((size_t)Nn * FFn * 2);
  unsigned* m32 = (unsigned*)alloc((size_t)Tn * MW * 4);  // 2 MB packed mask

  cvt_bf16<<<(Nn * Dn + 255) / 256, 256, 0, stream>>>(x, xb, Nn * Dn);
  cvt_bf16<<<64, 256, 0, stream>>>(Wq, wqb, 128 * 128);
  cvt_bf16<<<64, 256, 0, stream>>>(Wk, wkb, 128 * 128);
  cvt_bf16<<<64, 256, 0, stream>>>(Wv, wvb, 128 * 128);
  cvt_bf16<<<64, 256, 0, stream>>>(Wo, wob, 128 * 128);
  cvt_bf16<<<128, 256, 0, stream>>>(W1, w1b, 128 * 256);
  cvt_bf16<<<128, 256, 0, stream>>>(W2, w2b, 256 * 128);
  pack_mask<<<(Tn * MW) / 256, 256, 0, stream>>>(adj, m32);

  qkv_kernel<<<3072, 256, 0, stream>>>(xb, wqb, wkb, wvb, bq, bk, bv, Qh, Kt, Vh);
  attn_kernel<<<1024, 128, 0, stream>>>(Qh, Kt, Vh, m32, att);
  gemm_ln_kernel<4><<<1024, 256, 0, stream>>>(att, 128, wob, bo, x, ln1_g, ln1_b, x1, x1b);
  ff1_kernel<<<2048, 256, 0, stream>>>(x1b, w1b, b1, ffb);
  gemm_ln_kernel<8><<<1024, 256, 0, stream>>>(ffb, 256, w2b, b2, x1, ln2_g, ln2_b,
                                              (float*)d_out, nullptr);
}
